// HET_EglRelGraphConv_4793183503000
// MI455X (gfx1250) — matrix operand order repaired and checked by reading, not compiled, not run on hardware
//
#include <hip/hip_runtime.h>

typedef __attribute__((ext_vector_type(16))) __bf16 v16bf;
typedef __attribute__((ext_vector_type(16))) short  v16s;
typedef __attribute__((ext_vector_type(8)))  short  v8s;
typedef __attribute__((ext_vector_type(8)))  float  v8f;

#define N_NODES   20000
#define N_EDGES   320000
#define IN_FEAT   512
#define OUT_FEAT  512
#define NUM_RELS  32
#define NUM_BASES 8

#define TM 64      // edges per tile
#define TN 128     // output cols per block (4 n-tiles total)
#define KC 32      // K chunk (wmma 16x16x32)
#define MAX_TILES (N_EDGES / TM + NUM_RELS)   // 5032 upper bound

// LDS strides (ushorts), padded to dodge bank conflicts while keeping 16B align
#define A_STRIDE 72    // per buffer: 64*72*2 = 9216 B   (fragment-order rows)
#define B_STRIDE 40    // per buffer: 128*40*2 = 10240 B (transposed: [col][k])

// workspace layout (bytes)
#define WS_NTILES   0
#define WS_RELSTART 64
#define WS_TABLE    512                 // 5032 * 16 B
#define WS_XBF      131072              // 20000*512*2 = 20,480,000 B
#define WS_WT       20971520            // 32*512*512*2 = 16,777,216 B (transposed)

static __device__ __forceinline__ unsigned short f2bf(float f) {
  unsigned u = __float_as_uint(f);
  u += 0x7FFFu + ((u >> 16) & 1u);      // round-to-nearest-even
  return (unsigned short)(u >> 16);
}

union Frag16 { v16s s; v16bf b; };

static __device__ __forceinline__ v16s ld32B(const unsigned short* p) {
  const v8s* q = (const v8s*)p;         // 16B-aligned by construction
  v8s lo = q[0], hi = q[1];             // 2x ds_load_b128
  return __builtin_shufflevector(lo, hi, 0,1,2,3,4,5,6,7,8,9,10,11,12,13,14,15);
}

// ---------------------------------------------------------------------------
// 1a) x -> bf16 (once; GEMM kernel then moves pure bf16 bytes)
// ---------------------------------------------------------------------------
__global__ __launch_bounds__(256) void k_xcast(const float* __restrict__ x,
                                               unsigned short* __restrict__ xb) {
  int idx = blockIdx.x * 256 + threadIdx.x;
  if (idx < N_NODES * IN_FEAT) xb[idx] = f2bf(x[idx]);
}

// ---------------------------------------------------------------------------
// 1b) Wt[r][o][i] = sum_b w_comp[r,b] * bases[b][i][o]  (bf16, TRANSPOSED)
//     LDS transpose: reads coalesce along o, writes coalesce along i
// ---------------------------------------------------------------------------
__global__ __launch_bounds__(256) void k_weights(const float* __restrict__ bases,
                                                 const float* __restrict__ w_comp,
                                                 unsigned short* __restrict__ wt) {
  __shared__ float tile[32][33];
  const int r  = blockIdx.z;
  const int o0 = blockIdx.x * 32;
  const int i0 = blockIdx.y * 32;
  float wc[NUM_BASES];
#pragma unroll
  for (int b = 0; b < NUM_BASES; ++b) wc[b] = w_comp[r * NUM_BASES + b];

  const int tx = threadIdx.x & 31;
  const int ty = threadIdx.x >> 5;
  for (int ii = ty; ii < 32; ii += 8) {
    float acc = 0.f;
#pragma unroll
    for (int b = 0; b < NUM_BASES; ++b)
      acc = fmaf(wc[b], bases[b * (IN_FEAT * OUT_FEAT) + (i0 + ii) * OUT_FEAT + (o0 + tx)], acc);
    tile[ii][tx] = acc;
  }
  __syncthreads();
  for (int oo = ty; oo < 32; oo += 8)
    wt[((size_t)r << 18) + (size_t)(o0 + oo) * IN_FEAT + i0 + tx] = f2bf(tile[tx][oo]);
}

// ---------------------------------------------------------------------------
// 2) relation offsets (etype sorted) + tile table {edge_begin, edge_end, rel}
// ---------------------------------------------------------------------------
__global__ void k_tiles(const int* __restrict__ etype, int* __restrict__ rel_start,
                        int4* __restrict__ table, int* __restrict__ ntiles) {
  int t = threadIdx.x;
  if (t <= NUM_RELS) {
    int lo = 0, hi = N_EDGES;
    while (lo < hi) { int mid = (lo + hi) >> 1; if (etype[mid] < t) lo = mid + 1; else hi = mid; }
    rel_start[t] = lo;
  }
  __syncthreads();
  if (t == 0) {
    int cnt = 0;
    for (int r = 0; r < NUM_RELS; ++r) {
      int b = rel_start[r], e = rel_start[r + 1];
      for (int s = b; s < e; s += TM)
        table[cnt++] = make_int4(s, (s + TM < e) ? (s + TM) : e, r, 0);
    }
    *ntiles = cnt;
  }
}

// ---------------------------------------------------------------------------
// 3) out = broadcast(h_bias)
// ---------------------------------------------------------------------------
__global__ __launch_bounds__(256) void k_init(const float* __restrict__ h_bias,
                                              float* __restrict__ out) {
  int idx = blockIdx.x * 256 + threadIdx.x;
  if (idx < N_NODES * OUT_FEAT) out[idx] = h_bias[idx & (OUT_FEAT - 1)];
}

// ---------------------------------------------------------------------------
// 4) segmented edge-GEMM.  Block: 8 wave32, tile M=64 x N=128, K in 32-chunks.
//    Wave w: rows (w>>1)*16..+15, cols (w&1)*64..+63 -> 4 accums, 4 WMMA/step.
//    Double-buffered LDS + register prefetch: one barrier per K-step, global
//    latency hidden under the WMMA group, all fragment loads hoisted ahead of
//    the 4 back-to-back WMMAs.
// ---------------------------------------------------------------------------
__global__ __launch_bounds__(256) void k_rgcn(const unsigned short* __restrict__ xb,
                                              const float* __restrict__ norm,
                                              const int* __restrict__ src,
                                              const int* __restrict__ dst,
                                              const unsigned short* __restrict__ wt,
                                              const int4* __restrict__ table,
                                              const int* __restrict__ ntiles,
                                              float* __restrict__ out) {
  if ((int)blockIdx.x >= *ntiles) return;     // wave-uniform exit
  const int4 te = table[blockIdx.x];
  const int eb = te.x, ee = te.y, rel = te.z;
  const int n0 = blockIdx.y * TN;

  __shared__ __align__(16) unsigned short As[2][TM * A_STRIDE];
  __shared__ __align__(16) unsigned short Bt[2][TN * B_STRIDE];
  __shared__ int   sSrc[TM];
  __shared__ int   sDst[TM];
  __shared__ float sNrm[TM];

  const int tid   = threadIdx.x;
  const int lane  = tid & 31;
  const int wave  = tid >> 5;
  const int m_off = (wave >> 1) << 4;         // 0,16,32,48
  const int n_off = (wave & 1) << 6;          // 0,64

  if (tid < TM) {
    int e2 = eb + tid;
    if (e2 < ee) { sSrc[tid] = src[e2]; sDst[tid] = dst[e2]; sNrm[tid] = norm[e2]; }
    else         { sSrc[tid] = -1;      sDst[tid] = 0;       sNrm[tid] = 0.f; }
  }
  __syncthreads();

  // --- A staging: thread -> (row, K-group of 8); group g maps to frag slot ---
  const int  aRow   = tid >> 2;
  const int  aG     = tid & 3;                // K groups: 0-7,8-15,16-23,24-31
  const int  aOff   = aRow * A_STRIDE + aG * 8;
  const int  node   = sSrc[aRow];
  const bool aValid = (node >= 0);
  const unsigned short* xrow = xb + (size_t)(aValid ? node : 0) * IN_FEAT + aG * 8;

  // --- B staging: thread -> (col, K-half of 16); Wt row = output column ------
  const int bCol = tid >> 1;
  const int bHlf = tid & 1;
  const int bOff = bCol * B_STRIDE + bHlf * 16;
  const unsigned short* wrow = wt + ((size_t)rel << 18)
                             + (size_t)(n0 + bCol) * IN_FEAT + bHlf * 16;

  // --- per-lane fragment addresses (constant across K loop) ------------------
  const int ln    = lane & 15;
  const int aFrag = (m_off + ln) * A_STRIDE + (lane >> 4) * 16;
  const int kb    = (lane >> 4) * 16;

  // prologue: stage K-chunk 0 into buffer 0
  {
    v8s ra = *(const v8s*)xrow;
    if (!aValid) ra = v8s{};
    *(v8s*)(As[0] + aOff) = ra;
    const v8s* wp = (const v8s*)wrow;
    v8s* bp = (v8s*)(Bt[0] + bOff);
    bp[0] = wp[0];
    bp[1] = wp[1];
  }

  v8f acc[4] = {v8f{}, v8f{}, v8f{}, v8f{}};
  int p = 0;

  for (int k0 = 0; k0 < IN_FEAT; k0 += KC) {
    __syncthreads();                          // buffer p staged; buffer p^1 free

    // prefetch next K-chunk into registers (latency hides under WMMAs)
    const bool more = (k0 + KC) < IN_FEAT;    // uniform
    v8s ra{}, rb0{}, rb1{};
    if (more) {
      ra = *(const v8s*)(xrow + k0 + KC);
      if (!aValid) ra = v8s{};
      const v8s* wp = (const v8s*)(wrow + k0 + KC);
      rb0 = wp[0];
      rb1 = wp[1];
    }

    // hoist ALL fragment loads, then 4 back-to-back WMMAs
    Frag16 fa;  fa.s  = ld32B(As[p] + aFrag);
    Frag16 fb0; fb0.s = ld32B(Bt[p] + (n_off +  0 + ln) * B_STRIDE + kb);
    Frag16 fb1; fb1.s = ld32B(Bt[p] + (n_off + 16 + ln) * B_STRIDE + kb);
    Frag16 fb2; fb2.s = ld32B(Bt[p] + (n_off + 32 + ln) * B_STRIDE + kb);
    Frag16 fb3; fb3.s = ld32B(Bt[p] + (n_off + 48 + ln) * B_STRIDE + kb);

    acc[0] = __builtin_amdgcn_wmma_f32_16x16x32_bf16(false, fa.b, false, fb0.b,
                                                     (short)0, acc[0], false, false);
    acc[1] = __builtin_amdgcn_wmma_f32_16x16x32_bf16(false, fa.b, false, fb1.b,
                                                     (short)0, acc[1], false, false);
    acc[2] = __builtin_amdgcn_wmma_f32_16x16x32_bf16(false, fa.b, false, fb2.b,
                                                     (short)0, acc[2], false, false);
    acc[3] = __builtin_amdgcn_wmma_f32_16x16x32_bf16(false, fa.b, false, fb3.b,
                                                     (short)0, acc[3], false, false);

    // dump prefetched registers into the other buffer
    if (more) {
      *(v8s*)(As[p ^ 1] + aOff) = ra;
      v8s* bp = (v8s*)(Bt[p ^ 1] + bOff);
      bp[0] = rb0;
      bp[1] = rb1;
    }
    p ^= 1;
  }

  // --- epilogue: scale by per-edge norm (folded past GEMM), atomic scatter ---
  const int mb = m_off + ((lane & 16) ? 8 : 0);
#pragma unroll
  for (int j = 0; j < 8; ++j) {
    const int m = mb + j;
    if (eb + m < ee) {
      const float nrm = sNrm[m];
      float* op = out + (size_t)sDst[m] * OUT_FEAT + n0 + n_off + ln;
#pragma unroll
      for (int g = 0; g < 4; ++g)
        atomicAdd(op + g * 16, acc[g][j] * nrm);
    }
  }
}

// ---------------------------------------------------------------------------
extern "C" void kernel_launch(void* const* d_in, const int* in_sizes, int n_in,
                              void* d_out, int out_size, void* d_ws, size_t ws_size,
                              hipStream_t stream) {
  const float* x      = (const float*)d_in[0];
  const float* bases  = (const float*)d_in[1];
  const float* w_comp = (const float*)d_in[2];
  const float* h_bias = (const float*)d_in[3];
  const float* norm   = (const float*)d_in[4];
  const int*   src    = (const int*)d_in[5];
  const int*   dst    = (const int*)d_in[6];
  const int*   etype  = (const int*)d_in[7];
  float* out = (float*)d_out;

  char* ws = (char*)d_ws;
  int*            ntiles    = (int*)(ws + WS_NTILES);
  int*            rel_start = (int*)(ws + WS_RELSTART);
  int4*           table     = (int4*)(ws + WS_TABLE);
  unsigned short* xbf       = (unsigned short*)(ws + WS_XBF);
  unsigned short* wtb       = (unsigned short*)(ws + WS_WT);

  k_xcast<<<(N_NODES * IN_FEAT + 255) / 256, 256, 0, stream>>>(x, xbf);
  {
    dim3 g(OUT_FEAT / 32, IN_FEAT / 32, NUM_RELS);
    k_weights<<<g, 256, 0, stream>>>(bases, w_comp, wtb);
  }
  k_tiles<<<1, 64, 0, stream>>>(etype, rel_start, table, ntiles);
  k_init<<<(N_NODES * OUT_FEAT + 255) / 256, 256, 0, stream>>>(h_bias, out);

  dim3 grid(MAX_TILES, OUT_FEAT / TN);
  k_rgcn<<<grid, 256, 0, stream>>>(xbf, norm, src, dst, wtb, table, ntiles, out);
}
